// UNetRes_EnBorder_Patchwise_30468497998445
// MI455X (gfx1250) — matrix operand order repaired and checked by reading, not compiled, not run on hardware
//
#include <hip/hip_runtime.h>

// ---------------------------------------------------------------------------
// CDNA5 (gfx1250) implementation of the UNetRes denoiser.
// One WMMA implicit-GEMM 3x3-conv primitive drives the whole network:
//   - resblock conv1:  A = w1 (unfold order), +ReLU
//   - resblock conv2:  A = w2 kernel-flipped reshape, epilogue gg/cnt*acc + x
//   - head/down/up/tail convs: plain OIHW, optional gg/cnt scale + residual
// K is laid out kernel-position-major (k = r*Cin + i) so every index decode is
// a shift/mask (all Cin and H==W are powers of two). bf16 WMMA
// (v_wmma_f32_16x16x32_bf16), f32 accum; wave32; 8 waves/WG; 64x128 tile.
// Weight tiles are staged into LDS with the Tensor Data Mover when available.
// ---------------------------------------------------------------------------

typedef __bf16 bf16_t;
typedef __attribute__((ext_vector_type(16))) __bf16 v16bf;
typedef __attribute__((ext_vector_type(8)))  float  v8f;
typedef __attribute__((ext_vector_type(4)))  unsigned int v4u;
typedef __attribute__((ext_vector_type(4)))  int    v4i;
typedef __attribute__((ext_vector_type(8)))  int    v8i;

#if defined(__has_builtin)
#  if __has_builtin(__builtin_amdgcn_tensor_load_to_lds) && __has_builtin(__builtin_amdgcn_s_wait_tensorcnt)
#    define HAVE_TDM 1
#  endif
#endif
#ifndef HAVE_TDM
#  define HAVE_TDM 0
#endif

__device__ __forceinline__ unsigned short f2bf_bits(float f) {
  union { float f; unsigned u; } a; a.f = f;
  unsigned r = a.u + 0x7FFFu + ((a.u >> 16) & 1u);   // round-to-nearest-even
  return (unsigned short)(r >> 16);
}
__device__ __forceinline__ bf16_t f2bf(float f) {
  unsigned short h = f2bf_bits(f);
  return __builtin_bit_cast(bf16_t, h);
}

#if HAVE_TDM
// ---- TDM: DMA a rows x cols bf16 tile (row stride = strideElems) into LDS
__device__ __forceinline__ void tdm_load_tile(const bf16_t* gsrc, unsigned ldsAddr,
                                              int rows, int cols, int strideElems) {
  unsigned long long ga = (unsigned long long)(size_t)gsrc;
  v4u g0;
  g0[0] = 1u;                                       // count=1, user mode
  g0[1] = ldsAddr;                                  // lds_addr (bytes)
  g0[2] = (unsigned)(ga & 0xFFFFFFFFu);             // global_addr[31:0]
  g0[3] = (unsigned)((ga >> 32) & 0x01FFFFFFu) | (2u << 30);  // addr[56:32], type=2
  v8i g1;
  g1[0] = 0x00010000;                               // data_size=1 (2 bytes)
  g1[1] = (int)0xFFFF0000u;                         // tensor_dim0[15:0]=0xffff (huge)
  g1[2] = (int)0xFFFF7FFFu;                         // tensor_dim0 hi / tensor_dim1 lo
  g1[3] = 0x00007FFF | (cols << 16);                // tensor_dim1 hi / tile_dim0
  g1[4] = rows;                                     // tile_dim1 (tile_dim2=0)
  g1[5] = strideElems;                              // tensor_dim0_stride[31:0]
  g1[6] = 0;                                        // stride hi / dim1_stride lo
  g1[7] = 0;
  v4i gz = {0, 0, 0, 0};
#  if __clang_major__ >= 23
  v8i gz8 = {0, 0, 0, 0, 0, 0, 0, 0};
  __builtin_amdgcn_tensor_load_to_lds(g0, g1, gz, gz, gz8, 0);
#  else
  __builtin_amdgcn_tensor_load_to_lds(g0, g1, gz, gz, 0);
#  endif
}
#endif

// ---- weight conversion to K = r*Cin + i order (r = dy*3+dx), bf16, padded
// src: row-major OIHW flatten (M, Cin*9) i.e. w[m][i*9 + r]
__global__ void wconv_pad(const float* __restrict__ w, bf16_t* __restrict__ A,
                          int M, int Cin, int logCin, int Mpad, int Kpad) {
  int idx = blockIdx.x * 256 + threadIdx.x;
  if (idx >= Mpad * Kpad) return;
  int m = idx / Kpad, k = idx - m * Kpad;
  int r = k >> logCin, i = k & (Cin - 1);
  float v = (m < M && r < 9) ? w[(size_t)m * 9 * Cin + i * 9 + r] : 0.f;
  A[idx] = f2bf(v);
}

// ---- w2 (9C x C) -> conv-form: A[m][r*C+i] = w2[(m*9 + (8-r))*C + i]
__global__ void wconv_fold(const float* __restrict__ w2, bf16_t* __restrict__ A,
                           int C, int logC, int Mpad, int Kpad) {
  int idx = blockIdx.x * 256 + threadIdx.x;
  if (idx >= Mpad * Kpad) return;
  int m = idx / Kpad, k = idx - m * Kpad;
  int r = k >> logC, i = k & (C - 1);
  float v = (m < C && r < 9) ? w2[(size_t)(m * 9 + (8 - r)) * C + i] : 0.f;
  A[idx] = f2bf(v);
}

// ---------------------------------------------------------------------------
// Implicit-GEMM 3x3 conv (pad=1) with WMMA bf16.  H == W (powers of two).
//  A: bf16 [Mpad][Kpad] weights (k = r*Cin + i), X: f32 [2][Cin][H][W]
//  Y: f32 [2][M][H][W]; epilogue: relu? -> *gg/cnt(p,q) if gg>0 -> +R?
// ---------------------------------------------------------------------------
__global__ __launch_bounds__(256)
void conv3x3_wmma(const bf16_t* __restrict__ A,
                  const float*  __restrict__ X,
                  float*        __restrict__ Y,
                  const float*  __restrict__ R,
                  int Cin, int logCin, int M, int H, int W, int logW,
                  int Kpad, int relu, float gg) {
  __shared__ __align__(16) bf16_t As[64][32];
  __shared__ __align__(16) bf16_t Bs[128][32];

  const int tid  = threadIdx.x;
  const int lane = tid & 31;
  const int wave = tid >> 5;
  const int wm = (wave >> 1) * 16;
  const int wn = (wave & 1) * 64;

  const int mTile = blockIdx.y * 64;
  const int nTile = blockIdx.x * 128;
  const int logHW = 2 * logW;
  const int HW = 1 << logHW;

  v8f acc[4] = {};

  // B-gather coordinates: this thread fills Bs[nn][kk0 .. kk0+15]
  const int nn   = tid & 127;
  const int kk0  = (tid >> 7) * 16;
  const int npix = nTile + nn;
  const int bimg = npix >> logHW;
  const int rem  = npix & (HW - 1);
  const int py   = rem >> logW;
  const int px   = rem & (W - 1);
  const float* Xb = X + ((size_t)bimg << logCin << logHW);

#if !HAVE_TDM
  const int arow = tid >> 2;           // fallback A staging: 16B per thread
  const int acol = (tid & 3) * 8;
#endif

  for (int k0 = 0; k0 < Kpad; k0 += 32) {
    // ---- stage A tile (64x32 bf16) ----
#if HAVE_TDM
    if (wave == 0)
      tdm_load_tile(A + (size_t)mTile * Kpad + k0,
                    (unsigned)(size_t)&As[0][0], 64, 32, Kpad);
#else
    {
      const uint4* src = (const uint4*)(A + (size_t)(mTile + arow) * Kpad + k0 + acol);
      *(uint4*)&As[arow][acol] = *src;
      if (k0 + 32 < Kpad)
        __builtin_prefetch(A + (size_t)(mTile + arow) * Kpad + k0 + 32 + acol, 0, 1);
    }
#endif
    // ---- im2col gather B tile (128 x 32); all decode is shift/mask ----
#pragma unroll
    for (int t = 0; t < 16; t += 2) {
      unsigned pk = 0;
#pragma unroll
      for (int h = 0; h < 2; ++h) {
        int k = k0 + kk0 + t + h;
        int r = k >> logCin;
        float v = 0.f;
        if (r < 9) {                       // also rejects head's K padding
          int ci = k & (Cin - 1);
          int a  = r / 3;                  // r in [0,9): cheap
          int b  = r - a * 3;
          int yy = py + a - 1, xx = px + b - 1;
          if ((unsigned)yy < (unsigned)H && (unsigned)xx < (unsigned)W)
            v = Xb[((size_t)ci << logHW) + (yy << logW) + xx];
        }
        pk |= (unsigned)f2bf_bits(v) << (16 * h);
      }
      ((unsigned*)&Bs[nn][0])[(kk0 + t) >> 1] = pk;
    }
#if HAVE_TDM
    if (wave == 0) __builtin_amdgcn_s_wait_tensorcnt(0);
#endif
    __syncthreads();

    // ---- fragments per ISA 16-bit A/B layouts, issue WMMA ----
    union Uf { v16bf v; unsigned u[8]; };
    Uf af;
    {
      const int lm = lane & 15;
      const int kh = (lane >> 4) * 8;            // upper half-wave: K += 8
      const unsigned* Ar = (const unsigned*)&As[wm + lm][0];
#pragma unroll
      for (int p = 0; p < 8; ++p) {
        int kp = (p < 4 ? p * 2 : 16 + (p - 4) * 2) + kh;
        af.u[p] = Ar[kp >> 1];
      }
    }
    const int lnn = lane & 15;
  const int kh2 = (lane >> 4) * 4;             // uint offset: K += 8 for upper half-wave
#pragma unroll
    for (int j = 0; j < 4; ++j) {
      Uf bfr;
      const unsigned* Br = (const unsigned*)&Bs[wn + j * 16 + lnn][0];
#pragma unroll
      for (int p = 0; p < 8; ++p) bfr.u[p] = Br[kh2 + (p < 4 ? p : p + 4)];
      acc[j] = __builtin_amdgcn_wmma_f32_16x16x32_bf16(
                   false, af.v, false, bfr.v, (short)0, acc[j], false, false);
    }
    __syncthreads();
  }

  // ---- epilogue: relu / border count-scale / residual, NCHW store ----
  const int mBase = mTile + wm + ((lane >> 4) ? 8 : 0);
#pragma unroll
  for (int j = 0; j < 4; ++j) {
    int n  = nTile + wn + j * 16 + (lane & 15);
    int b2 = n >> logHW;
    int r2 = n & (HW - 1);
    int p2 = r2 >> logW;
    int q2 = r2 & (W - 1);
    float s = 1.f;
    if (gg > 0.f) {
      float ch = (p2 == 0 || p2 == H - 1) ? 2.f : 3.f;
      float cw = (q2 == 0 || q2 == W - 1) ? 2.f : 3.f;
      s = gg / (ch * cw);
    }
#pragma unroll
    for (int rr = 0; rr < 8; ++rr) {
      int m = mBase + rr;
      if (m < M) {
        float v = acc[j][rr];
        if (relu) v = v > 0.f ? v : 0.f;
        v *= s;
        size_t idx = ((size_t)b2 * M + m) * HW + r2;
        if (R) v += R[idx];
        Y[idx] = v;
      }
    }
  }
}

// ---------------------------------------------------------------------------
// elementwise helpers
// ---------------------------------------------------------------------------
__global__ void makein4(const float* __restrict__ x0, const float* __restrict__ sigma,
                        float* __restrict__ out, int H, int W) {
  int idx = blockIdx.x * 256 + threadIdx.x;
  if (idx >= 2 * 4 * H * W) return;
  int x = idx % W; int t = idx / W;
  int y = t % H;   t /= H;
  int c = t % 4;   int b = t / 4;
  out[idx] = (c < 3) ? x0[((size_t)(b * 3 + c) * H + y) * W + x] : sigma[b];
}

__global__ void unshuffle_k(const float* __restrict__ in, float* __restrict__ out,
                            int C, int H, int W) {   // [2,C,H,W] -> [2,4C,H/2,W/2]
  int idx = blockIdx.x * 256 + threadIdx.x;
  if (idx >= 2 * C * H * W) return;
  int Ho = H / 2, Wo = W / 2, C4 = 4 * C;
  int x  = idx % Wo; int t = idx / Wo;
  int y  = t % Ho;   t /= Ho;
  int c4 = t % C4;   int b = t / C4;
  int c = c4 >> 2, s = (c4 >> 1) & 1, u = c4 & 1;
  out[idx] = in[((size_t)(b * C + c) * H + 2 * y + s) * W + 2 * x + u];
}

__global__ void shuffle_k(const float* __restrict__ in, float* __restrict__ out,
                          int C, int H, int W) {     // [2,C,H,W] -> [2,C/4,2H,2W]
  int idx = blockIdx.x * 256 + threadIdx.x;
  if (idx >= 2 * C * H * W) return;
  int Co = C / 4, Ho = 2 * H, Wo = 2 * W;
  int x2 = idx % Wo; int t = idx / Wo;
  int y2 = t % Ho;   t /= Ho;
  int c  = t % Co;   int b = t / Co;
  int ci = c * 4 + (y2 & 1) * 2 + (x2 & 1);
  out[idx] = in[((size_t)(b * C + ci) * H + (y2 >> 1)) * W + (x2 >> 1)];
}

// ---------------------------------------------------------------------------
extern "C" void kernel_launch(void* const* d_in, const int* in_sizes, int n_in,
                              void* d_out, int out_size, void* d_ws, size_t ws_size,
                              hipStream_t stream) {
  (void)out_size; (void)ws_size;

  // ---- identify inputs by flat size (order of equal sizes is stable under
  //      both insertion-order and alphabetical pytree flattening) ----
  const float *x0 = nullptr, *sigma = nullptr, *head = nullptr, *tail = nullptr;
  const float *down[3] = {}, *up[3] = {};
  const float *body[7][4][2] = {};
  int c36 = 0, c147 = 0, c589 = 0, c235 = 0, c18 = 0, c73 = 0, c294 = 0;
  for (int i = 0; i < n_in; ++i) {
    const float* p = (const float*)d_in[i];
    switch (in_sizes[i]) {
      case 98304:  x0 = p; break;
      case 2:      sigma = p; break;
      case 2304:   head = p; break;
      case 1728:   tail = p; break;
      case 18432:  (c18++  == 0 ? down[0] : up[0]) = p; break;
      case 73728:  (c73++  == 0 ? down[1] : up[1]) = p; break;
      case 294912: (c294++ == 0 ? down[2] : up[2]) = p; break;
      case 36864:  { int t = c36++;  int bi = (t < 8) ? 0 : 6; int r = t & 7; body[bi][r >> 1][r & 1] = p; } break;
      case 147456: { int t = c147++; int bi = (t < 8) ? 1 : 5; int r = t & 7; body[bi][r >> 1][r & 1] = p; } break;
      case 589824: { int t = c589++; int bi = (t < 8) ? 2 : 4; int r = t & 7; body[bi][r >> 1][r & 1] = p; } break;
      case 2359296:{ int t = c235++; body[3][t >> 1][t & 1] = p; } break;
      default: break;
    }
  }

  // ---- workspace carve ----
  char* base = (char*)d_ws;
  bf16_t* wslot = (bf16_t*)base;            base += (size_t)512 * 4608 * 2;   // 4.5 MB
  float* in4  = (float*)base;               base += (size_t)131072  * 4;
  float* bufX = (float*)base;               base += (size_t)2097152 * 4;
  float* bufT = (float*)base;               base += (size_t)2097152 * 4;
  float* x1b  = (float*)base;               base += (size_t)2097152 * 4;
  float* x2b  = (float*)base;               base += (size_t)1048576 * 4;
  float* x3b  = (float*)base;               base += (size_t)524288  * 4;
  float* tmpS = (float*)base;               base += (size_t)1048576 * 4;

  auto ggf = [](int T) { double v = (3.0 * T - 2.0) / T; return (float)(v * v); };
  const float gg1 = ggf(128), gg2 = ggf(64), gg3 = ggf(32), gg4 = ggf(16);
  auto lg = [](int v) { int l = 0; while ((1 << l) < v) ++l; return l; };

  auto loadw = [&](const float* wp, int Mo, int Cin) {
    int Kpad = (9 * Cin + 31) / 32 * 32, Mpad = (Mo + 63) / 64 * 64;
    int total = Mpad * Kpad;
    wconv_pad<<<dim3((total + 255) / 256), dim3(256), 0, stream>>>(
        wp, wslot, Mo, Cin, lg(Cin), Mpad, Kpad);
  };
  auto loadw2 = [&](const float* wp, int C) {
    int Kpad = 9 * C, Mpad = (C + 63) / 64 * 64;
    int total = Mpad * Kpad;
    wconv_fold<<<dim3((total + 255) / 256), dim3(256), 0, stream>>>(
        wp, wslot, C, lg(C), Mpad, Kpad);
  };
  auto conv = [&](const float* Xp, int Cin, float* Yp, int Mo, int Hs, int Ws,
                  const float* Rp, int relu, float gg) {
    int Kpad = (9 * Cin + 31) / 32 * 32, Mpad = (Mo + 63) / 64 * 64;
    dim3 grid((2 * Hs * Ws) / 128, Mpad / 64, 1);
    conv3x3_wmma<<<grid, dim3(256), 0, stream>>>(
        wslot, Xp, Yp, Rp, Cin, lg(Cin), Mo, Hs, Ws, lg(Ws), Kpad, relu, gg);
  };
  auto resblock = [&](float* xp, int C, int Hs, int Ws, const float* w1,
                      const float* w2, float gg) {
    loadw(w1, C, C);
    conv(xp, C, bufT, C, Hs, Ws, nullptr, 1, 0.f);     // h = relu(W1 * unfold(x))
    loadw2(w2, C);
    conv(bufT, C, xp, C, Hs, Ws, xp, 0, gg);           // x = x + gg/cnt*fold(W2*h)
  };
  auto bodyrun = [&](int bi, float* xp, int C, int Hs, int Ws, float gg) {
    for (int k = 0; k < 4; ++k) resblock(xp, C, Hs, Ws, body[bi][k][0], body[bi][k][1], gg);
  };
  auto ew = [](int total) { return dim3((total + 255) / 256); };

  // ---- encoder ----
  makein4<<<ew(131072), 256, 0, stream>>>(x0, sigma, in4, 128, 128);
  loadw(head, 64, 4);
  conv(in4, 4, bufX, 64, 128, 128, nullptr, 0, 0.f);
  bodyrun(0, bufX, 64, 128, 128, gg1);
  hipMemcpyAsync(x1b, bufX, (size_t)2097152 * 4, hipMemcpyDeviceToDevice, stream);

  loadw(down[0], 32, 64);
  conv(bufX, 64, tmpS, 32, 128, 128, nullptr, 0, 0.f);
  unshuffle_k<<<ew(2 * 32 * 128 * 128), 256, 0, stream>>>(tmpS, bufX, 32, 128, 128);
  bodyrun(1, bufX, 128, 64, 64, gg2);
  hipMemcpyAsync(x2b, bufX, (size_t)1048576 * 4, hipMemcpyDeviceToDevice, stream);

  loadw(down[1], 64, 128);
  conv(bufX, 128, tmpS, 64, 64, 64, nullptr, 0, 0.f);
  unshuffle_k<<<ew(2 * 64 * 64 * 64), 256, 0, stream>>>(tmpS, bufX, 64, 64, 64);
  bodyrun(2, bufX, 256, 32, 32, gg3);
  hipMemcpyAsync(x3b, bufX, (size_t)524288 * 4, hipMemcpyDeviceToDevice, stream);

  loadw(down[2], 128, 256);
  conv(bufX, 256, tmpS, 128, 32, 32, nullptr, 0, 0.f);
  unshuffle_k<<<ew(2 * 128 * 32 * 32), 256, 0, stream>>>(tmpS, bufX, 128, 32, 32);
  bodyrun(3, bufX, 512, 16, 16, gg4);

  // ---- decoder ----
  shuffle_k<<<ew(2 * 512 * 16 * 16), 256, 0, stream>>>(bufX, tmpS, 512, 16, 16);
  loadw(up[2], 256, 128);
  conv(tmpS, 128, bufX, 256, 32, 32, x3b, 0, gg3);
  bodyrun(4, bufX, 256, 32, 32, gg3);

  shuffle_k<<<ew(2 * 256 * 32 * 32), 256, 0, stream>>>(bufX, tmpS, 256, 32, 32);
  loadw(up[1], 128, 64);
  conv(tmpS, 64, bufX, 128, 64, 64, x2b, 0, gg2);
  bodyrun(5, bufX, 128, 64, 64, gg2);

  shuffle_k<<<ew(2 * 128 * 64 * 64), 256, 0, stream>>>(bufX, tmpS, 128, 64, 64);
  loadw(up[0], 64, 32);
  conv(tmpS, 32, bufX, 64, 128, 128, x1b, 0, gg1);
  bodyrun(6, bufX, 64, 128, 128, gg1);

  loadw(tail, 3, 64);
  conv(bufX, 64, (float*)d_out, 3, 128, 128, x0, 0, gg1);
}
